// PatternBranch_6846177870564
// MI455X (gfx1250) — matrix operand order repaired and checked by reading, not compiled, not run on hardware
//
#include <hip/hip_runtime.h>
#include <math.h>

typedef __attribute__((ext_vector_type(16))) _Float16 v16h;
typedef __attribute__((ext_vector_type(8)))  float    v8f;

#define BATCH 256
#define IN_H 64
#define CIN 3
#define OUT_H 32
#define COUT 128
#define KDIM 27                 // 3*3*3 im2col depth, padded to 32
#define TILE_PIX 16             // output pixels per block
#define TILES_PER_SAMPLE 64     // 32*32 / 16

// Fused: conv3x3 s2 SAME (+bias,+relu) via WMMA, then in-LDS reductions for
// match score, pattern logit (gathered channels) and 3 base logits.
// Each block: 1 sample x 16 output pixels x all 128 channels (8 waves x 1 WMMA).
__global__ __launch_bounds__(256)
void conv_fused_kernel(const float* __restrict__ inputs,   // [256,64,64,3]
                       const float* __restrict__ conv_w,   // [3,3,3,128]
                       const float* __restrict__ conv_b,   // [128]
                       const float* __restrict__ match_w,  // [128]
                       const float* __restrict__ pat_w,    // [32*32*32,1]
                       const float* __restrict__ base_w,   // [32*32*128,3]
                       const int*   __restrict__ psi,      // [32]
                       float* __restrict__ partials)       // [B*64][8]
{
    __shared__ _Float16 a_lds[TILE_PIX * 32];   // im2col A tile (f16)
    __shared__ _Float16 b_lds[32 * COUT];       // weight B tile  (f16)
    __shared__ float    feats[TILE_PIX * COUT]; // conv output tile (f32)
    __shared__ int      psi_lds[32];
    __shared__ float    red[8 * 5];

    const int t    = threadIdx.x;
    const int blk  = blockIdx.x;
    const int b    = blk / TILES_PER_SAMPLE;
    const int tile = blk % TILES_PER_SAMPLE;

    // ---- stage A: im2col patch, f32 -> f16.  SAME pad for s2/k3/in64: pad_lo=0, pad_hi=1
    for (int i = 0; i < 2; ++i) {
        int idx = t + 256 * i;              // 16*32 = 512 elems
        int m = idx >> 5, k = idx & 31;
        float v = 0.f;
        if (k < KDIM) {
            int kh = k / 9, kw = (k % 9) / 3, ci = k % 3;
            int mg = tile * TILE_PIX + m;
            int oh = mg >> 5, ow = mg & 31;
            int ih = 2 * oh + kh, iw = 2 * ow + kw;
            if (ih < IN_H && iw < IN_H)
                v = inputs[((b * IN_H + ih) * IN_H + iw) * CIN + ci];
        }
        a_lds[idx] = (_Float16)v;
    }
    // ---- stage B: conv weights [k][c], f16 (L2-resident, 13.8KB)
    for (int i = 0; i < 16; ++i) {
        int idx = t + 256 * i;              // 32*128 = 4096 elems
        int k = idx >> 7, c = idx & 127;
        float w = (k < KDIM) ? conv_w[k * COUT + c] : 0.f;
        b_lds[idx] = (_Float16)w;
    }
    if (t < 32) psi_lds[t] = psi[t];
    __syncthreads();

    // ---- WMMA: wave w -> 16 pixels x 16 channels (chan base = w*16)
    const int wave = t >> 5;
    const int lane = t & 31;
    const int half = lane >> 4;     // lane group 0..15 vs 16..31
    const int mrow = lane & 15;     // A row (pixel within tile)
    const int chan = wave * 16 + (lane & 15);

    union { v16h v; _Float16 e[16]; } Af, Bf;
    // A 16x32 f16, contiguous k order chosen here: row = lane%16; K 0-15 (lanes<16) / 16-31 (lanes>=16).
    #pragma unroll
    for (int e = 0; e < 16; ++e) Af.e[e] = a_lds[mrow * 32 + (half * 16 + e)];
    // B 32x16 f16, same contiguous k order as A: col = lane%16; K 0-15 (lanes<16) / 16-31 (lanes>=16).
    #pragma unroll
    for (int e = 0; e < 16; ++e) Bf.e[e] = b_lds[(half * 16 + e) * COUT + chan];

    v8f acc = {};
    acc = __builtin_amdgcn_wmma_f32_16x16x32_f16(
        /*neg_a=*/false, Af.v, /*neg_b=*/false, Bf.v,
        /*c_mod=*/(short)0, acc, /*reuse_a=*/false, /*reuse_b=*/false);

    // C layout: row = half*8 + r, col = lane%16.  Bias + ReLU -> LDS f32.
    const float bias = conv_b[chan];
    union { v8f v; float e[8]; } Cf; Cf.v = acc;
    #pragma unroll
    for (int r = 0; r < 8; ++r) {
        float v = Cf.e[r] + bias;
        feats[(half * 8 + r) * COUT + chan] = v > 0.f ? v : 0.f;
    }
    __syncthreads();

    // ---- fused reductions over the 16x128 tile
    float mAcc = 0.f, pAcc = 0.f, b0 = 0.f, b1 = 0.f, b2 = 0.f;
    for (int i = 0; i < 8; ++i) {
        int idx = t + 256 * i;              // 2048 (m, c) pairs
        int m = idx >> 7, c = idx & 127;
        float v = feats[idx];
        mAcc += v * match_w[c];
        int row = (tile * TILE_PIX + m) * COUT + c;   // base_w row (pix*128 + c)
        b0 += v * base_w[row * 3 + 0];
        b1 += v * base_w[row * 3 + 1];
        b2 += v * base_w[row * 3 + 2];
    }
    for (int i = 0; i < 2; ++i) {
        int pp = t + 256 * i;               // 16*32 = 512 (m, kk) pairs
        int m = pp >> 5, kk = pp & 31;
        int c = psi_lds[kk];
        pAcc += feats[m * COUT + c] * pat_w[(tile * TILE_PIX + m) * 32 + kk];
    }

    // ---- block reduce 5 scalars: wave shuffle tree, then cross-wave via LDS
    float vals[5] = { mAcc, pAcc, b0, b1, b2 };
    #pragma unroll
    for (int v = 0; v < 5; ++v) {
        float s = vals[v];
        #pragma unroll
        for (int msk = 16; msk >= 1; msk >>= 1) s += __shfl_xor(s, msk, 32);
        if (lane == 0) red[wave * 5 + v] = s;
    }
    __syncthreads();
    if (t < 5) {
        float s = 0.f;
        #pragma unroll
        for (int w = 0; w < 8; ++w) s += red[w * 5 + t];
        partials[blk * 8 + t] = s;          // deterministic: one writer per slot
    }
}

// Per-sample finalize: GAP->match, sigmoid pattern prob, softmax base, routing.
__global__ __launch_bounds__(256)
void finalize_kernel(const float* __restrict__ partials,
                     const float* __restrict__ match_b,
                     const float* __restrict__ pat_b,
                     const float* __restrict__ base_b,
                     float* __restrict__ out)   // [256,3]
{
    int s = blockIdx.x * blockDim.x + threadIdx.x;
    if (s >= BATCH) return;
    float a0 = 0.f, a1 = 0.f, a2 = 0.f, a3 = 0.f, a4 = 0.f;
    for (int t2 = 0; t2 < TILES_PER_SAMPLE; ++t2) {
        const float* p = partials + (s * TILES_PER_SAMPLE + t2) * 8;
        a0 += p[0]; a1 += p[1]; a2 += p[2]; a3 += p[3]; a4 += p[4];
    }
    bool match = (a0 * (1.f / 1024.f) + match_b[0]) > 0.f;
    float p = 1.f / (1.f + expf(-(a1 + pat_b[0])));
    float l0 = a2 + base_b[0], l1 = a3 + base_b[1], l2 = a4 + base_b[2];
    float mx = fmaxf(l0, fmaxf(l1, l2));
    float e0 = expf(l0 - mx), e1 = expf(l1 - mx), e2 = expf(l2 - mx);
    float inv = 1.f / (e0 + e1 + e2);
    float o0, o1, o2;
    if (match && p >= 0.5f) { o0 = p; o1 = (1.f - p) * 0.5f; o2 = o1; }
    else                    { o0 = e0 * inv; o1 = e1 * inv; o2 = e2 * inv; }
    out[s * 3 + 0] = o0;
    out[s * 3 + 1] = o1;
    out[s * 3 + 2] = o2;
}

extern "C" void kernel_launch(void* const* d_in, const int* in_sizes, int n_in,
                              void* d_out, int out_size, void* d_ws, size_t ws_size,
                              hipStream_t stream) {
    const float* inputs  = (const float*)d_in[0];
    const float* conv_w  = (const float*)d_in[1];
    const float* conv_b  = (const float*)d_in[2];
    const float* match_w = (const float*)d_in[3];
    const float* match_b = (const float*)d_in[4];
    const float* pat_w   = (const float*)d_in[5];
    const float* pat_b   = (const float*)d_in[6];
    const float* base_w  = (const float*)d_in[7];
    const float* base_b  = (const float*)d_in[8];
    const int*   psi     = (const int*)d_in[9];

    float* partials = (float*)d_ws;   // BATCH*64*8 floats = 512 KB

    conv_fused_kernel<<<BATCH * TILES_PER_SAMPLE, 256, 0, stream>>>(
        inputs, conv_w, conv_b, match_w, pat_w, base_w, psi, partials);
    finalize_kernel<<<1, 256, 0, stream>>>(
        partials, match_b, pat_b, base_b, (float*)d_out);
}
